// LSTMModel_5634997092554
// MI455X (gfx1250) — matrix operand order repaired and checked by reading, not compiled, not run on hardware
//
#include <hip/hip_runtime.h>

typedef __attribute__((ext_vector_type(16))) _Float16 v16h;
typedef __attribute__((ext_vector_type(8)))  float    v8f;

#define Tlen 512
#define Hd   64
#define G4   256   // 4*H
#define TB   16    // batch rows per workgroup

#define L2E  1.44269504088896340736f   // log2(e)

__device__ __forceinline__ float fexp2(float x) { return __builtin_amdgcn_exp2f(x); }
__device__ __forceinline__ float frcp(float x)  { return __builtin_amdgcn_rcpf(x); }

// sigmoid(x) = 1 / (1 + exp(-x)) ; branch-free, native v_exp_f32 + v_rcp_f32
__device__ __forceinline__ float sigf(float x) {
    return frcp(1.0f + fexp2(-L2E * x));
}

// tanh: native v_tanh_f32 when the builtin exists, else branch-free formula
#if defined(__has_builtin)
#  if __has_builtin(__builtin_amdgcn_tanhf)
#    define HAVE_NATIVE_TANH 1
#  endif
#endif
__device__ __forceinline__ float tanh_fast(float x) {
#ifdef HAVE_NATIVE_TANH
    return __builtin_amdgcn_tanhf(x);
#else
    float e = fexp2(2.0f * L2E * x);
    return 1.0f - 2.0f * frcp(e + 1.0f);
#endif
}

#define WMMA_F16(A, B, C) \
  __builtin_amdgcn_wmma_f32_16x16x32_f16(false, (A), false, (B), (short)0, (C), false, false)

__global__ __launch_bounds__(256, 1)
void lstm2_fused_kernel(const float* __restrict__ x,
                        const float* __restrict__ Wih0, const float* __restrict__ Whh0,
                        const float* __restrict__ bih0, const float* __restrict__ bhh0,
                        const float* __restrict__ Wih1, const float* __restrict__ Whh1,
                        const float* __restrict__ bih1, const float* __restrict__ bhh1,
                        const float* __restrict__ fcW,  const float* __restrict__ fcb,
                        float* __restrict__ out)
{
    // -------- LDS state (~58 KB) --------
    __shared__ float xs[TB][Tlen];                       // input tile for this batch slab
    __shared__ float Gs[TB][G4];                         // gate pre-activations (reused L0/L1)
    __shared__ __align__(32) _Float16 hA0[2][32][16];    // layer0 h, WMMA A layout (K=64)
    __shared__ __align__(32) _Float16 hA1[4][32][16];    // layer1 A = [h0_t | h1_prev] (K=128)
    __shared__ float h2f[TB][Hd];                        // last-step layer1 h (f32) for FC

    const int tid  = threadIdx.x;
    const int lane = tid & 31;
    const int wave = tid >> 5;
    const int b0   = blockIdx.x * TB;

    // -------- one-time preload --------
    for (int idx = tid; idx < TB * Tlen; idx += 256) {
        int bb = idx >> 9, t = idx & (Tlen - 1);
        xs[bb][t] = x[(size_t)(b0 + bb) * Tlen + t];
    }
    for (int idx = tid; idx < 2 * 32 * 16; idx += 256) ((_Float16*)hA0)[idx] = (_Float16)0.0f;
    for (int idx = tid; idx < 4 * 32 * 16; idx += 256) ((_Float16*)hA1)[idx] = (_Float16)0.0f;

    // -------- weight B-fragments in VGPRs (per-wave: 2 gate-column tiles) --------
    // B-layout (16-bit, 32x16 KxN): lanes 0-15 -> N=0..15 / K=0..7,16..23,
    // lanes 16-31 -> same N / K=8..15,24..31 (table order, matches A scatter).
    const int n0   = (wave * 2) * 16;      // gate column base, tile 0
    const int n1   = (wave * 2 + 1) * 16;  // gate column base, tile 1
    const int bn   = lane & 15;
    const int kb16 = (lane >> 4) * 8;

    v16h B0[2][2];   // [ntile][kchunk]  layer0:  K=64  from Whh0^T
    v16h B1[2][4];   // [ntile][kchunk]  layer1:  K=128 from [Wih1^T ; Whh1^T]
    #pragma unroll
    for (int c = 0; c < 2; ++c) {
        #pragma unroll
        for (int i = 0; i < 16; ++i) {
            int k = c * 32 + kb16 + (i & 7) + ((i >> 3) << 4);
            B0[0][c][i]     = (_Float16)Whh0[(n0 + bn) * Hd + k];
            B0[1][c][i]     = (_Float16)Whh0[(n1 + bn) * Hd + k];
            B1[0][c][i]     = (_Float16)Wih1[(n0 + bn) * Hd + k];
            B1[1][c][i]     = (_Float16)Wih1[(n1 + bn) * Hd + k];
            B1[0][c + 2][i] = (_Float16)Whh1[(n0 + bn) * Hd + k];
            B1[1][c + 2][i] = (_Float16)Whh1[(n1 + bn) * Hd + k];
        }
    }

    // -------- per-thread gate constants in registers (loop-invariant) --------
    const int gb = tid >> 4;   // batch row handled in gate phase
    const int gj = tid & 15;   // hidden-column base (stride 16, 4 per thread)

    float w0r[16], b0r[16], b1r[16];   // [u*4 + gate], gate: 0=i,1=f,2=g,3=o
    #pragma unroll
    for (int u = 0; u < 4; ++u) {
        #pragma unroll
        for (int q = 0; q < 4; ++q) {
            int col = gj + u * 16 + q * 64;
            w0r[u * 4 + q] = Wih0[col];                   // Wih0 is [256][1]
            b0r[u * 4 + q] = bih0[col] + bhh0[col];
            b1r[u * 4 + q] = bih1[col] + bhh1[col];
        }
    }
    float c0r[4] = {0.f, 0.f, 0.f, 0.f};   // cell state, layer 0 (thread-private)
    float c1r[4] = {0.f, 0.f, 0.f, 0.f};   // cell state, layer 1

    __syncthreads();

    const int m0 = (lane >> 4) * 8;   // C-layout row base for this lane
    const int cc = lane & 15;         // C-layout column for this lane

    for (int t = 0; t < Tlen; ++t) {
        // ======== layer 0: G = h0_{t-1} @ Whh0^T ========
        {
            v16h a0 = *(const v16h*)&hA0[0][lane][0];
            v16h a1 = *(const v16h*)&hA0[1][lane][0];
            v8f acc0 = {}, acc1 = {};
            acc0 = WMMA_F16(a0, B0[0][0], acc0);
            acc0 = WMMA_F16(a1, B0[0][1], acc0);
            acc1 = WMMA_F16(a0, B0[1][0], acc1);
            acc1 = WMMA_F16(a1, B0[1][1], acc1);
            #pragma unroll
            for (int r = 0; r < 8; ++r) {
                Gs[m0 + r][n0 + cc] = acc0[r];
                Gs[m0 + r][n1 + cc] = acc1[r];
            }
        }
        __syncthreads();

        // ======== layer 0 gates: fold in x_t * Wih0 + bias ========
        {
            float xv = xs[gb][t];
            #pragma unroll
            for (int u = 0; u < 4; ++u) {
                int j = gj + u * 16;
                float pi = Gs[gb][j      ] + xv * w0r[u * 4 + 0] + b0r[u * 4 + 0];
                float pf = Gs[gb][j +  64] + xv * w0r[u * 4 + 1] + b0r[u * 4 + 1];
                float pg = Gs[gb][j + 128] + xv * w0r[u * 4 + 2] + b0r[u * 4 + 2];
                float po = Gs[gb][j + 192] + xv * w0r[u * 4 + 3] + b0r[u * 4 + 3];
                float ig = sigf(pi), fg = sigf(pf), gg = tanh_fast(pg), og = sigf(po);
                float c = fg * c0r[u] + ig * gg;
                c0r[u] = c;
                float h = og * tanh_fast(c);
                // scatter h into WMMA A layout
                int kk = j & 31, chunk = j >> 5;
                int ln = gb + (((kk >> 3) & 1) << 4);
                int ix = (kk & 7) + ((kk >> 4) << 3);
                _Float16 hh = (_Float16)h;
                hA0[chunk][ln][ix] = hh;       // next-step layer0 recurrent input
                hA1[chunk][ln][ix] = hh;       // this-step layer1 feed-forward input
            }
        }
        __syncthreads();

        // ======== layer 1: G = [h0_t | h1prev] @ [Wih1^T ; Whh1^T]  (K=128) ========
        {
            v16h a0 = *(const v16h*)&hA1[0][lane][0];
            v16h a1 = *(const v16h*)&hA1[1][lane][0];
            v16h a2 = *(const v16h*)&hA1[2][lane][0];
            v16h a3 = *(const v16h*)&hA1[3][lane][0];
            v8f acc0 = {}, acc1 = {};
            acc0 = WMMA_F16(a0, B1[0][0], acc0);
            acc0 = WMMA_F16(a1, B1[0][1], acc0);
            acc0 = WMMA_F16(a2, B1[0][2], acc0);
            acc0 = WMMA_F16(a3, B1[0][3], acc0);
            acc1 = WMMA_F16(a0, B1[1][0], acc1);
            acc1 = WMMA_F16(a1, B1[1][1], acc1);
            acc1 = WMMA_F16(a2, B1[1][2], acc1);
            acc1 = WMMA_F16(a3, B1[1][3], acc1);
            #pragma unroll
            for (int r = 0; r < 8; ++r) {
                Gs[m0 + r][n0 + cc] = acc0[r];
                Gs[m0 + r][n1 + cc] = acc1[r];
            }
        }
        __syncthreads();

        // ======== layer 1 gates ========
        {
            #pragma unroll
            for (int u = 0; u < 4; ++u) {
                int j = gj + u * 16;
                float pi = Gs[gb][j      ] + b1r[u * 4 + 0];
                float pf = Gs[gb][j +  64] + b1r[u * 4 + 1];
                float pg = Gs[gb][j + 128] + b1r[u * 4 + 2];
                float po = Gs[gb][j + 192] + b1r[u * 4 + 3];
                float ig = sigf(pi), fg = sigf(pf), gg = tanh_fast(pg), og = sigf(po);
                float c = fg * c1r[u] + ig * gg;
                c1r[u] = c;
                float h = og * tanh_fast(c);
                int kk = j & 31, chunk = 2 + (j >> 5);
                int ln = gb + (((kk >> 3) & 1) << 4);
                int ix = (kk & 7) + ((kk >> 4) << 3);
                hA1[chunk][ln][ix] = (_Float16)h;  // next-step layer1 recurrent input
                if (t == Tlen - 1) h2f[gb][j] = h;
            }
        }
        __syncthreads();
    }

    // ======== final FC: out[b] = h2[T-1,b,:] . fcW + fcb ========
    if (tid < TB) {
        float s = fcb[0];
        #pragma unroll
        for (int j = 0; j < Hd; ++j) s += h2f[tid][j] * fcW[j];
        out[b0 + tid] = s;
    }
}

extern "C" void kernel_launch(void* const* d_in, const int* in_sizes, int n_in,
                              void* d_out, int out_size, void* d_ws, size_t ws_size,
                              hipStream_t stream)
{
    const float* x    = (const float*)d_in[0];
    const float* Wih0 = (const float*)d_in[1];
    const float* Whh0 = (const float*)d_in[2];
    const float* bih0 = (const float*)d_in[3];
    const float* bhh0 = (const float*)d_in[4];
    const float* Wih1 = (const float*)d_in[5];
    const float* Whh1 = (const float*)d_in[6];
    const float* bih1 = (const float*)d_in[7];
    const float* bhh1 = (const float*)d_in[8];
    const float* fcW  = (const float*)d_in[9];
    const float* fcb  = (const float*)d_in[10];
    float* out = (float*)d_out;

    const int B = 2048;
    lstm2_fused_kernel<<<dim3(B / TB), dim3(256), 0, stream>>>(
        x, Wih0, Whh0, bih0, bhh0, Wih1, Whh1, bih1, bhh1, fcW, fcb, out);
}
